// LSTM_61143154426126
// MI455X (gfx1250) — matrix operand order repaired and checked by reading, not compiled, not run on hardware
//
#include <hip/hip_runtime.h>

// Problem constants (from reference: B=1024, T=600, F=2, H=128)
#define Tt 600
#define Bb 1024
#define Hh 128
#define Gg 512       // 4*H gate columns
#define MTILE 16     // batch rows per workgroup (WMMA M)
#define NTHR 256     // 8 waves (wave32)

typedef __attribute__((ext_vector_type(16))) __bf16       v16bf;
typedef __attribute__((ext_vector_type(8)))  float        v8f;
typedef __attribute__((ext_vector_type(4)))  unsigned int v4u;

union Frag { v16bf bf; v4u u[2]; };

__device__ inline unsigned short f2bf(float f) {
  unsigned int u = __float_as_uint(f);
  u += 0x7FFFu + ((u >> 16) & 1u);          // round-to-nearest-even
  return (unsigned short)(u >> 16);
}

__device__ inline float fast_sigmoid(float x) {
  float e = __builtin_amdgcn_exp2f(x * -1.442695040888963f);
  return __builtin_amdgcn_rcpf(1.0f + e);
}
__device__ inline float fast_tanh(float x) {
  float e = __builtin_amdgcn_exp2f(x * -2.885390081777926f);
  return 2.0f * __builtin_amdgcn_rcpf(1.0f + e) - 1.0f;
}

// A-fragment (16x32 bf16, row-major [16][128] bf16 buffer in LDS).
// lane<16: M=lane, v0-3=K(k0..k0+7), v4-7=K(k0+16..k0+23)
// lane>=16: M=lane-16, v0-3=K(k0+8..k0+15), v4-7=K(k0+24..k0+31)
__device__ inline v16bf ldA(const unsigned short* buf, int lane, int k0) {
  const unsigned char* p = (const unsigned char*)buf
      + (lane & 15) * (Hh * 2) + (k0 + ((lane >> 4) << 3)) * 2;
  Frag f;
  f.u[0] = *(const v4u*)(p);
  f.u[1] = *(const v4u*)(p + 32);
  return f.bf;
}

// B-fragment (32x16 bf16) from weight matrix W[512][128] bf16 row-major.
// B[k][n] = W[nbase+n][k].  lane<16: N=lane, v0-3=K(k0..k0+7), v4-7=K(k0+16..k0+23); lane>=16: K+=8.
__device__ inline v16bf ldB(const unsigned short* w, int lane, int nbase, int k0) {
  const unsigned char* p = (const unsigned char*)w
      + (nbase + (lane & 15)) * (Hh * 2) + k0 * 2 + ((lane >> 4) << 4);
  Frag f;
  f.u[0] = *(const v4u*)(p);
  f.u[1] = *(const v4u*)(p + 32);
  return f.bf;
}

// fp32 global weights -> bf16 LDS (row-major preserved)
__device__ inline void cvt_store_weights(unsigned short* dst, const float* src,
                                         int nelem, int tid) {
  const float4* s4 = (const float4*)src;
  uint2* d2 = (uint2*)dst;
  int n4 = nelem >> 2;
  for (int i = tid; i < n4; i += NTHR) {
    float4 v = s4[i];
    uint2 pk;
    pk.x = (unsigned)f2bf(v.x) | ((unsigned)f2bf(v.y) << 16);
    pk.y = (unsigned)f2bf(v.z) | ((unsigned)f2bf(v.w) << 16);
    d2[i] = pk;
  }
}

// ---------------- Layer 0: x[B,T,2] -> h1[T,B,128] (bf16 in d_ws) ------------
__global__ void __launch_bounds__(NTHR, 1)
lstm_layer0(const float* __restrict__ x,
            const float* __restrict__ w_ih0,
            const float* __restrict__ b_ih0,
            const float* __restrict__ b_hh0,
            const float* __restrict__ w_hh0,
            unsigned short* __restrict__ h1g) {
  extern __shared__ unsigned char smem[];
  unsigned short* wlds  = (unsigned short*)smem;                 // 512*128 bf16
  unsigned short* h0buf = (unsigned short*)(smem + Gg * Hh * 2); // 16*128 bf16

  const int tid  = threadIdx.x;
  const int lane = tid & 31;
  const int wv   = tid >> 5;
  const int JC   = wv * 16;       // this wave's 16 columns of H
  const int ln   = lane & 15;
  const int hi   = lane >> 4;
  const int b0   = blockIdx.x * MTILE;

  cvt_store_weights(wlds, w_hh0, Gg * Hh, tid);
  v4u z = {0u, 0u, 0u, 0u};
  ((v4u*)h0buf)[tid] = z;   // h_{-1} = 0

  // Per-lane constants: combined bias and input weights for (gate g, col JC+ln)
  float bsum[4], wx0[4], wx1[4];
#pragma unroll
  for (int g = 0; g < 4; ++g) {
    int n = g * Hh + JC + ln;
    bsum[g] = b_ih0[n] + b_hh0[n];
    wx0[g]  = w_ih0[n * 2 + 0];
    wx1[g]  = w_ih0[n * 2 + 1];
  }

  __syncthreads();

  // Hoist all recurrent-weight B-fragments into registers (loop-invariant):
  // 4 K-blocks x 4 gates x 8 VGPRs = 128 VGPRs.
  v16bf wb[4][4];
#pragma unroll
  for (int k = 0; k < 4; ++k)
#pragma unroll
    for (int g = 0; g < 4; ++g)
      wb[k][g] = ldB(wlds, lane, g * Hh + JC, k * 32);

  v8f cst = {};             // c state: rows (reg+8*hi), col JC+ln

  for (int t = 0; t < Tt; ++t) {
    v8f acc[4];
    // gates = b + x_t @ w_ih0^T  (F=2 -> pure VALU)
#pragma unroll
    for (int r = 0; r < 8; ++r) {
      int row = r + (hi << 3);
      const float* xp = x + ((size_t)(b0 + row) * Tt + t) * 2;
      float x0 = xp[0], x1 = xp[1];
#pragma unroll
      for (int g = 0; g < 4; ++g)
        acc[g][r] = bsum[g] + wx0[g] * x0 + wx1[g] * x1;
    }
    // gates += h_{t-1} @ w_hh0^T : load all A-frags, then 16 back-to-back WMMAs
    v16bf a[4];
#pragma unroll
    for (int k = 0; k < 4; ++k) a[k] = ldA(h0buf, lane, k * 32);
#pragma unroll
    for (int k = 0; k < 4; ++k)
#pragma unroll
      for (int g = 0; g < 4; ++g)
        acc[g] = __builtin_amdgcn_wmma_f32_16x16x32_bf16(
            false, a[k], false, wb[k][g], (short)0, acc[g], false, false);

    __syncthreads();        // everyone done reading h0buf
#pragma unroll
    for (int r = 0; r < 8; ++r) {
      float ig = fast_sigmoid(acc[0][r]);
      float fg = fast_sigmoid(acc[1][r]);
      float gg = fast_tanh(acc[2][r]);
      float og = fast_sigmoid(acc[3][r]);
      float c  = fg * cst[r] + ig * gg;
      cst[r]   = c;
      float h  = og * fast_tanh(c);
      int row  = r + (hi << 3);
      h0buf[row * Hh + JC + ln] = f2bf(h);
    }
    __syncthreads();        // h_t complete in LDS
    // coalesced bf16 tile store: h1g[t][b0..b0+15][0..127]
    v4u hv = ((const v4u*)h0buf)[tid];
    ((v4u*)(h1g + ((size_t)t * Bb + b0) * Hh))[tid] = hv;
  }
}

// -------- Layer 1 + FC: h1[T,B,128] bf16 -> out[B,T,2] f32 -------------------
__global__ void __launch_bounds__(NTHR, 1)
lstm_layer1_fc(const unsigned short* __restrict__ h1g,
               const float* __restrict__ w_ih1,
               const float* __restrict__ w_hh1,
               const float* __restrict__ b_ih1,
               const float* __restrict__ b_hh1,
               const float* __restrict__ fc_w,
               const float* __restrict__ fc_b,
               float* __restrict__ out) {
  extern __shared__ unsigned char smem[];
  unsigned short* wi    = (unsigned short*)smem;                       // 128KB
  unsigned short* wh    = (unsigned short*)(smem + (size_t)Gg * Hh * 2);
  unsigned short* h2buf = (unsigned short*)(smem + (size_t)2 * Gg * Hh * 2);
  unsigned short* xb0   = h2buf + MTILE * Hh;
  unsigned short* xb1   = xb0 + MTILE * Hh;
  float*          fcwL  = (float*)(xb1 + MTILE * Hh);                  // 2*128 f32

  const int tid  = threadIdx.x;
  const int lane = tid & 31;
  const int wv   = tid >> 5;
  const int JC   = wv * 16;
  const int ln   = lane & 15;
  const int hi   = lane >> 4;
  const int b0   = blockIdx.x * MTILE;

  cvt_store_weights(wi, w_ih1, Gg * Hh, tid);
  cvt_store_weights(wh, w_hh1, Gg * Hh, tid);
  fcwL[tid] = fc_w[tid];                       // 256 floats, 256 threads
  v4u z = {0u, 0u, 0u, 0u};
  ((v4u*)h2buf)[tid] = z;
  // stage input tile t=0
  ((v4u*)xb0)[tid] = ((const v4u*)(h1g + (size_t)b0 * Hh))[tid];

  float bsum[4];
#pragma unroll
  for (int g = 0; g < 4; ++g) {
    int n = g * Hh + JC + ln;
    bsum[g] = b_ih1[n] + b_hh1[n];
  }
  float fcb = fc_b[lane & 1];

  __syncthreads();

  // Register-resident B-fragments for the RECURRENT weights only (the serial
  // critical path); the x-projection streams its B-frags from LDS since it has
  // no step-to-step dependency. 128 VGPRs.
  v16bf whb[4][4];
#pragma unroll
  for (int k = 0; k < 4; ++k)
#pragma unroll
    for (int g = 0; g < 4; ++g)
      whb[k][g] = ldB(wh, lane, g * Hh + JC, k * 32);

  v8f cst = {};

  for (int t = 0; t < Tt; ++t) {
    unsigned short* xc = (t & 1) ? xb1 : xb0;
    unsigned short* xn = (t & 1) ? xb0 : xb1;
    // prefetch next input tile into registers (latency hidden under WMMAs)
    int tn = (t + 1 < Tt) ? (t + 1) : t;
    v4u nxt = ((const v4u*)(h1g + ((size_t)tn * Bb + b0) * Hh))[tid];

    v8f acc[4];
#pragma unroll
    for (int g = 0; g < 4; ++g)
#pragma unroll
      for (int r = 0; r < 8; ++r) acc[g][r] = bsum[g];

    // gates += x_t @ w_ih1^T   (B-frags streamed from LDS)
#pragma unroll
    for (int k0 = 0; k0 < Hh; k0 += 32) {
      v16bf a = ldA(xc, lane, k0);
#pragma unroll
      for (int g = 0; g < 4; ++g) {
        v16bf b = ldB(wi, lane, g * Hh + JC, k0);
        acc[g] = __builtin_amdgcn_wmma_f32_16x16x32_bf16(
            false, a, false, b, (short)0, acc[g], false, false);
      }
    }
    // gates += h2_{t-1} @ w_hh1^T  (register-resident weights, dense WMMA chain)
    v16bf ah[4];
#pragma unroll
    for (int k = 0; k < 4; ++k) ah[k] = ldA(h2buf, lane, k * 32);
#pragma unroll
    for (int k = 0; k < 4; ++k)
#pragma unroll
      for (int g = 0; g < 4; ++g)
        acc[g] = __builtin_amdgcn_wmma_f32_16x16x32_bf16(
            false, ah[k], false, whb[k][g], (short)0, acc[g], false, false);

    __syncthreads();        // reads of h2buf / xc complete
#pragma unroll
    for (int r = 0; r < 8; ++r) {
      float ig = fast_sigmoid(acc[0][r]);
      float fg = fast_sigmoid(acc[1][r]);
      float gg = fast_tanh(acc[2][r]);
      float og = fast_sigmoid(acc[3][r]);
      float c  = fg * cst[r] + ig * gg;
      cst[r]   = c;
      float h  = og * fast_tanh(c);
      int row  = r + (hi << 3);
      h2buf[row * Hh + JC + ln] = f2bf(h);
    }
    ((v4u*)xn)[tid] = nxt;  // stage next input tile
    __syncthreads();        // h2_t and next tile visible
    // FC on wave 0: out[b0+row][t][o] = h2_t[row] . fc_w[o] + fc_b[o]
    // (runs in the barrier shadow of the next step's matmuls)
    if (wv == 0) {
      int row = lane >> 1, o = lane & 1;
      float s = fcb;
      const v4u* hr = (const v4u*)(h2buf + row * Hh);
      const v4u* wr = (const v4u*)(fcwL + o * Hh);
#pragma unroll
      for (int kb = 0; kb < 16; ++kb) {
        v4u hv = hr[kb];
        v4u w0 = wr[kb * 2], w1 = wr[kb * 2 + 1];
        s += __uint_as_float(hv.x << 16)          * __uint_as_float(w0.x);
        s += __uint_as_float(hv.x & 0xffff0000u)  * __uint_as_float(w0.y);
        s += __uint_as_float(hv.y << 16)          * __uint_as_float(w0.z);
        s += __uint_as_float(hv.y & 0xffff0000u)  * __uint_as_float(w0.w);
        s += __uint_as_float(hv.z << 16)          * __uint_as_float(w1.x);
        s += __uint_as_float(hv.z & 0xffff0000u)  * __uint_as_float(w1.y);
        s += __uint_as_float(hv.w << 16)          * __uint_as_float(w1.z);
        s += __uint_as_float(hv.w & 0xffff0000u)  * __uint_as_float(w1.w);
      }
      out[((size_t)(b0 + row) * Tt + t) * 2 + o] = s;
    }
  }
}

extern "C" void kernel_launch(void* const* d_in, const int* in_sizes, int n_in,
                              void* d_out, int out_size, void* d_ws, size_t ws_size,
                              hipStream_t stream) {
  (void)in_sizes; (void)n_in; (void)out_size; (void)ws_size;
  const float* x     = (const float*)d_in[0];
  const float* w_ih0 = (const float*)d_in[1];
  const float* w_hh0 = (const float*)d_in[2];
  const float* b_ih0 = (const float*)d_in[3];
  const float* b_hh0 = (const float*)d_in[4];
  const float* w_ih1 = (const float*)d_in[5];
  const float* w_hh1 = (const float*)d_in[6];
  const float* b_ih1 = (const float*)d_in[7];
  const float* b_hh1 = (const float*)d_in[8];
  const float* fc_w  = (const float*)d_in[9];
  const float* fc_b  = (const float*)d_in[10];
  float* out = (float*)d_out;
  unsigned short* h1g = (unsigned short*)d_ws;   // [T][B][H] bf16, 150MiB

  size_t smemA = (size_t)Gg * Hh * 2 + (size_t)MTILE * Hh * 2;            // ~132KB
  size_t smemB = (size_t)2 * Gg * Hh * 2 + (size_t)3 * MTILE * Hh * 2
               + (size_t)2 * Hh * 4;                                      // ~269KB

  lstm_layer0<<<dim3(Bb / MTILE), dim3(NTHR), smemA, stream>>>(
      x, w_ih0, b_ih0, b_hh0, w_hh0, h1g);
  lstm_layer1_fc<<<dim3(Bb / MTILE), dim3(NTHR), smemB, stream>>>(
      h1g, w_ih1, w_hh1, b_ih1, b_hh1, fc_w, fc_b, out);
}
